// DLO_net_single_2602750181912
// MI455X (gfx1250) — compile-verified, matrix operand order checked by reading, not run on hardware
//
#include <hip/hip_runtime.h>
#include <hip/hip_bf16.h>
#include <cstdint>

typedef __attribute__((ext_vector_type(8)))  __bf16 v8bf;
typedef __attribute__((ext_vector_type(16))) __bf16 v16bf;
typedef __attribute__((ext_vector_type(8)))  float  v8f;

#define BATCH 2
#define NPTS  8192
#define MPTS  4096
#define DFEAT 512
#define CORR_SCALE 0.04419417382415922f   // 1/sqrt(512)

__device__ inline __bf16 f32_to_bf16(float x) {   // round-to-nearest-even
  uint32_t u = __builtin_bit_cast(uint32_t, x);
  uint32_t r = u + 0x7FFFu + ((u >> 16) & 1u);
  uint16_t hb = (uint16_t)(r >> 16);
  return __builtin_bit_cast(__bf16, hb);
}
__device__ inline float bf16_to_f32(__bf16 h) {
  uint32_t u = ((uint32_t)__builtin_bit_cast(uint16_t, h)) << 16;
  return __builtin_bit_cast(float, u);
}

// ---------------- kernel 1: fp32 -> bf16 hi/lo split (one-time) ----------------
__global__ __launch_bounds__(256)
void split_bf16_kernel(const float* __restrict__ src,
                       __bf16* __restrict__ hi, __bf16* __restrict__ lo, int count) {
  int i = blockIdx.x * blockDim.x + threadIdx.x;
  if (i >= count) return;
  float x  = src[i];
  __bf16 h = f32_to_bf16(x);
  __bf16 l = f32_to_bf16(x - bf16_to_f32(h));
  hi[i] = h; lo[i] = l;
}

// -------- kernel 2: fused correlation GEMM (bf16x3 WMMA) + online softmax ------
__global__ __launch_bounds__(256)
void corr_softmax_kernel(const __bf16* __restrict__ f0h, const __bf16* __restrict__ f0l,
                         const __bf16* __restrict__ f1h, const __bf16* __restrict__ f1l,
                         const float*  __restrict__ pts1,
                         float* __restrict__ out_corr,   // [b,n]
                         float* __restrict__ out_cp1t2,  // [b,n,3]
                         float* __restrict__ val_max_ws) // [b,n]
{
  __shared__ __align__(16) __bf16 sAh[16 * DFEAT];
  __shared__ __align__(16) __bf16 sAl[16 * DFEAT];
  __shared__ float pM[8][16], pS[8][16], pW[8][16];

  const int batch  = blockIdx.y;
  const int rowBlk = blockIdx.x;          // 16-row block of n
  const int tid  = threadIdx.x;
  const int wv   = tid >> 5;              // wave id 0..7
  const int lane = tid & 31;
  const int r    = lane & 15;             // A row within tile / B column N
  const int h    = lane >> 4;             // half-wave

  // stage A block (16 rows x 512 K) hi/lo into LDS (contiguous, layout-identical)
  {
    const uint32_t* gh = (const uint32_t*)(f0h + ((size_t)batch * NPTS + rowBlk * 16) * DFEAT);
    const uint32_t* gl = (const uint32_t*)(f0l + ((size_t)batch * NPTS + rowBlk * 16) * DFEAT);
    uint32_t* lh = (uint32_t*)sAh;
    uint32_t* ll = (uint32_t*)sAl;
    for (int i = tid; i < 16 * DFEAT / 2; i += 256) { lh[i] = gh[i]; ll[i] = gl[i]; }
  }
  __syncthreads();

  // per-lane online-softmax stats for rows (v + 8h) of this block
  float Mx[8], Sx[8], Wx[8];
  #pragma unroll
  for (int v = 0; v < 8; ++v) { Mx[v] = -INFINITY; Sx[v] = 0.f; Wx[v] = 0.f; }

  const __bf16* ahRow = sAh + r * DFEAT;
  const __bf16* alRow = sAl + r * DFEAT;

  for (int t = wv; t < MPTS / 16; t += 8) {            // 32 m-tiles per wave
    v8f acc = {0.f,0.f,0.f,0.f,0.f,0.f,0.f,0.f};
    const __bf16* bhRow = f1h + ((size_t)batch * MPTS + t * 16 + r) * DFEAT;
    const __bf16* blRow = f1l + ((size_t)batch * MPTS + t * 16 + r) * DFEAT;

    for (int c = 0; c < DFEAT / 32; ++c) {             // 16 K-chunks of 32
      const int k0 = c * 32 + h * 8;                   // ISA 16-bit A 16x32 layout
      v16bf aH = __builtin_shufflevector(*(const v8bf*)(ahRow + k0),
                                         *(const v8bf*)(ahRow + k0 + 16),
                                         0,1,2,3,4,5,6,7,8,9,10,11,12,13,14,15);
      v16bf aL = __builtin_shufflevector(*(const v8bf*)(alRow + k0),
                                         *(const v8bf*)(alRow + k0 + 16),
                                         0,1,2,3,4,5,6,7,8,9,10,11,12,13,14,15);
      v16bf bH = __builtin_shufflevector(*(const v8bf*)(bhRow + k0),
                                         *(const v8bf*)(bhRow + k0 + 16),
                                         0,1,2,3,4,5,6,7,8,9,10,11,12,13,14,15);
      v16bf bL = __builtin_shufflevector(*(const v8bf*)(blRow + k0),
                                         *(const v8bf*)(blRow + k0 + 16),
                                         0,1,2,3,4,5,6,7,8,9,10,11,12,13,14,15);
      // bf16x3 fp32-recovery: hi*hi + hi*lo + lo*hi, f32 accumulate
      acc = __builtin_amdgcn_wmma_f32_16x16x32_bf16(false, aH, false, bH, (short)0, acc, false, false);
      acc = __builtin_amdgcn_wmma_f32_16x16x32_bf16(false, aH, false, bL, (short)0, acc, false, false);
      acc = __builtin_amdgcn_wmma_f32_16x16x32_bf16(false, aL, false, bH, (short)0, acc, false, false);
    }

    // online softmax update; lane's column index j = t*16 + r
    const float jcol = (float)(t * 16 + r);
    #pragma unroll
    for (int v = 0; v < 8; ++v) {                      // row = v + 8h
      float x = acc[v] * CORR_SCALE;
      float tmax = x;
      tmax = fmaxf(tmax, __shfl_xor(tmax, 1, 32));
      tmax = fmaxf(tmax, __shfl_xor(tmax, 2, 32));
      tmax = fmaxf(tmax, __shfl_xor(tmax, 4, 32));
      tmax = fmaxf(tmax, __shfl_xor(tmax, 8, 32));
      float newM = fmaxf(Mx[v], tmax);
      float rescale = __expf(Mx[v] - newM);            // 0 when Mx == -inf
      float e  = __expf(x - newM);
      float es = e, ew = e * jcol;
      es += __shfl_xor(es, 1, 32); ew += __shfl_xor(ew, 1, 32);
      es += __shfl_xor(es, 2, 32); ew += __shfl_xor(ew, 2, 32);
      es += __shfl_xor(es, 4, 32); ew += __shfl_xor(ew, 4, 32);
      es += __shfl_xor(es, 8, 32); ew += __shfl_xor(ew, 8, 32);
      Sx[v] = Sx[v] * rescale + es;
      Wx[v] = Wx[v] * rescale + ew;
      Mx[v] = newM;
    }
  }

  // publish per-wave partials (stats are allreduced across 16-lane groups)
  if (r == 0) {
    #pragma unroll
    for (int v = 0; v < 8; ++v) {
      int row = v + 8 * h;
      pM[wv][row] = Mx[v]; pS[wv][row] = Sx[v]; pW[wv][row] = Wx[v];
    }
  }
  __syncthreads();

  // merge 8 wave-partials per row (log-sum-exp merge) and write epilogue
  if (tid < 16) {
    float bigM = -INFINITY;
    #pragma unroll
    for (int w = 0; w < 8; ++w) bigM = fmaxf(bigM, pM[w][tid]);
    float S = 0.f, W = 0.f;
    #pragma unroll
    for (int w = 0; w < 8; ++w) {
      float f = __expf(pM[w][tid] - bigM);
      S += pS[w][tid] * f;
      W += pW[w][tid] * f;
    }
    size_t oidx = (size_t)batch * NPTS + rowBlk * 16 + tid;
    float corres = W / S;
    out_corr[oidx]   = corres;
    val_max_ws[oidx] = 1.0f / S;                       // max(prob) = exp(0)/Z
    int ci = (int)corres;                              // trunc like .astype(int32)
    ci = ci < 0 ? 0 : (ci > MPTS - 1 ? MPTS - 1 : ci);
    const float* p1 = pts1 + ((size_t)batch * MPTS + ci) * 3;
    out_cp1t2[oidx * 3 + 0] = p1[0];
    out_cp1t2[oidx * 3 + 1] = p1[1];
    out_cp1t2[oidx * 3 + 2] = p1[2];
  }
}

// ------------- kernel 3: per-batch top-k via LDS bitonic sort + gather ----------
__global__ __launch_bounds__(1024)
void topk_kernel(const float* __restrict__ val_max, const float* __restrict__ pts0,
                 float* __restrict__ out_valm, float* __restrict__ out_indm,
                 float* __restrict__ out_cppts0) {
  __shared__ uint32_t key[NPTS];   // ascending key == descending value (vals > 0)
  __shared__ uint32_t idx[NPTS];
  const int b   = blockIdx.x;
  const int tid = threadIdx.x;

  for (int i = tid; i < NPTS; i += 1024) {
    uint32_t fb = __builtin_bit_cast(uint32_t, val_max[(size_t)b * NPTS + i]);
    key[i] = 0xFFFFFFFFu - fb;
    idx[i] = (uint32_t)i;
  }
  __syncthreads();

  for (int k = 2; k <= NPTS; k <<= 1) {
    for (int j = k >> 1; j > 0; j >>= 1) {
      for (int i = tid; i < NPTS; i += 1024) {
        int ixj = i ^ j;
        if (ixj > i) {
          bool up = ((i & k) == 0);
          uint32_t ka = key[i], kb = key[ixj];
          uint32_t ia = idx[i], ib = idx[ixj];
          bool agtb = (ka > kb) || (ka == kb && ia > ib);  // tie: lower index first
          if (agtb == up) { key[i] = kb; key[ixj] = ka; idx[i] = ib; idx[ixj] = ia; }
        }
      }
      __syncthreads();
    }
  }

  for (int i = tid; i < MPTS; i += 1024) {
    uint32_t id = idx[i];
    out_valm[(size_t)b * MPTS + i] = __builtin_bit_cast(float, 0xFFFFFFFFu - key[i]);
    out_indm[(size_t)b * MPTS + i] = (float)id;
    const float* sp = pts0 + ((size_t)b * NPTS + id) * 3;
    float* dp = out_cppts0 + ((size_t)b * MPTS + i) * 3;
    dp[0] = sp[0]; dp[1] = sp[1]; dp[2] = sp[2];
  }
}

extern "C" void kernel_launch(void* const* d_in, const int* in_sizes, int n_in,
                              void* d_out, int out_size, void* d_ws, size_t ws_size,
                              hipStream_t stream) {
  (void)in_sizes; (void)n_in; (void)out_size; (void)ws_size;
  const float* feat0 = (const float*)d_in[0];
  const float* feat1 = (const float*)d_in[1];
  const float* pts0  = (const float*)d_in[2];
  const float* pts1  = (const float*)d_in[3];

  // workspace layout (bytes): f0h 16M | f0l 16M | f1h 8M | f1l 8M | val_max 64K
  char* ws = (char*)d_ws;
  __bf16* f0h = (__bf16*)(ws);
  __bf16* f0l = (__bf16*)(ws + 16777216);
  __bf16* f1h = (__bf16*)(ws + 33554432);
  __bf16* f1l = (__bf16*)(ws + 41943040);
  float*  vmx = (float* )(ws + 50331648);

  float* out        = (float*)d_out;
  float* out_corr   = out;             // [2,8192]      16384
  float* out_cp1t2  = out + 16384;     // [2,8192,3]    49152
  float* out_cppts0 = out + 65536;     // [2,4096,3]    24576
  float* out_valm   = out + 90112;     // [2,4096]       8192
  float* out_indm   = out + 98304;     // [2,4096]       8192 (indices as float)

  const int c0 = BATCH * NPTS * DFEAT;   // 8388608
  const int c1 = BATCH * MPTS * DFEAT;   // 4194304
  split_bf16_kernel<<<(c0 + 255) / 256, 256, 0, stream>>>(feat0, f0h, f0l, c0);
  split_bf16_kernel<<<(c1 + 255) / 256, 256, 0, stream>>>(feat1, f1h, f1l, c1);

  dim3 grid(NPTS / 16, BATCH);           // 512 x 2 workgroups, 8 waves each
  corr_softmax_kernel<<<grid, 256, 0, stream>>>(f0h, f0l, f1h, f1l, pts1,
                                                out_corr, out_cp1t2, vmx);
  topk_kernel<<<BATCH, 1024, 0, stream>>>(vmx, pts0, out_valm, out_indm, out_cppts0);
}
